// SimCLR_Loss_60911226192410
// MI455X (gfx1250) — matrix operand order repaired and checked by reading, not compiled, not run on hardware
//
#include <hip/hip_runtime.h>
#include <hip/hip_bf16.h>
#include <math.h>

typedef _Float16 v16h __attribute__((ext_vector_type(16)));
typedef float    v8f  __attribute__((ext_vector_type(8)));

#define B_ROWS 4096
#define D_DIM  128
#define N_ROWS 8192
#define INV_T  2.0f          // 1 / TEMPERATURE

#define CHUNK_COLS 64                      // columns staged per LDS chunk
#define COL_STRIDE 136                     // halves: 272B padded stride (17*16B) -> no bank conflicts
#define NCHUNK     (N_ROWS / CHUNK_COLS)   // 128 chunks

// ---------------------------------------------------------------------------
// Kernel 1: row-normalize z = [z_i ; z_j], store as f16 into workspace.
// ---------------------------------------------------------------------------
__global__ void __launch_bounds__(256)
normalize_kernel(const float* __restrict__ zi, const float* __restrict__ zj,
                 _Float16* __restrict__ znh) {
    const int wave = (blockIdx.x * blockDim.x + threadIdx.x) >> 5;
    const int lane = threadIdx.x & 31;
    if (wave >= N_ROWS) return;
    const float* src = (wave < B_ROWS)
                         ? (zi + (size_t)wave * D_DIM)
                         : (zj + (size_t)(wave - B_ROWS) * D_DIM);
    float4 v = *(const float4*)(src + lane * 4);
    float ss = v.x * v.x + v.y * v.y + v.z * v.z + v.w * v.w;
    #pragma unroll
    for (int m = 16; m >= 1; m >>= 1) ss += __shfl_xor(ss, m, 32);
    const float scale = 1.0f / fmaxf(sqrtf(ss), 1e-8f);
    _Float16 h[4];
    h[0] = (_Float16)(v.x * scale);
    h[1] = (_Float16)(v.y * scale);
    h[2] = (_Float16)(v.z * scale);
    h[3] = (_Float16)(v.w * scale);
    *(float2*)(znh + (size_t)wave * D_DIM + lane * 4) = *(float2*)h;
}

// ---------------------------------------------------------------------------
// Kernel 2: positive logits pos[i] = dot(zn[i], zn[i +/- B]) / T.
// ---------------------------------------------------------------------------
__global__ void __launch_bounds__(256)
pos_kernel(const _Float16* __restrict__ znh, float* __restrict__ pos) {
    const int wave = (blockIdx.x * blockDim.x + threadIdx.x) >> 5;
    const int lane = threadIdx.x & 31;
    if (wave >= N_ROWS) return;
    const int partner = (wave < B_ROWS) ? (wave + B_ROWS) : (wave - B_ROWS);
    const _Float16* a = znh + (size_t)wave    * D_DIM + lane * 4;
    const _Float16* b = znh + (size_t)partner * D_DIM + lane * 4;
    float s = 0.0f;
    #pragma unroll
    for (int t = 0; t < 4; ++t) s += (float)a[t] * (float)b[t];
    #pragma unroll
    for (int m = 16; m >= 1; m >>= 1) s += __shfl_xor(s, m, 32);
    if (lane == 0) pos[wave] = s * INV_T;
}

// ---------------------------------------------------------------------------
// Kernel 3: LDS-staged, double-buffered Gram sweep + streaming logsumexp.
// 8 waves/block, 16 rows per wave (128 rows/block), grid = 64 blocks.
// Column chunks of 64 are DMA'd into LDS with global_load_async_to_lds_b128
// (ASYNCcnt) while the previous chunk is consumed via ds_load -> v_wmma.
// ---------------------------------------------------------------------------
__global__ void __launch_bounds__(256)
gram_rowloss_kernel(const _Float16* __restrict__ znh,
                    const float* __restrict__ pos,
                    float* __restrict__ rowloss) {
    __shared__ _Float16 tile[2][CHUNK_COLS * COL_STRIDE];  // 2 x 17408 B

    const int tid     = threadIdx.x;
    const int lane    = tid & 31;
    const int wv      = tid >> 5;              // 0..7
    const int r0      = blockIdx.x * 128 + wv * 16;
    const int halfsel = lane >> 4;             // 0: lanes 0-15, 1: lanes 16-31
    const int l15     = lane & 15;

    // --- A fragments: rows r0..r0+15, 16-bit A 16x32 layout per ISA:
    // lane<16 holds K {kb+0..7, kb+16..23}; lane>=16 holds K {kb+8..15, kb+24..31}
    v16h afrag[4];
    {
        const _Float16* arow = znh + (size_t)(r0 + l15) * D_DIM;
        #pragma unroll
        for (int kc = 0; kc < 4; ++kc) {
            const int kb = kc * 32 + halfsel * 8;
            v16h a;
            *(float4*)&a       = *(const float4*)(arow + kb);
            *((float4*)&a + 1) = *(const float4*)(arow + kb + 16);
            afrag[kc] = a;
        }
    }

    float acc[8];
    #pragma unroll
    for (int v = 0; v < 8; ++v) acc[v] = 0.0f;

    // Async cooperative stage of one 64-column chunk into LDS buffer b.
    // 64 cols x 256B = 1024 x 16B units; 256 threads -> 4 async b128 each.
    auto issue_chunk = [&](int cc, int b) {
        const char* gbase = (const char*)(znh + (size_t)cc * CHUNK_COLS * D_DIM);
        #pragma unroll
        for (int u = 0; u < 4; ++u) {
            const int unit = tid + u * 256;    // 0..1023
            const int c    = unit >> 4;        // column 0..63
            const int k16  = unit & 15;        // 16B unit within the column
            // Low 32 bits of the flat LDS-aperture address == LDS byte offset.
            uint32_t ldsoff = (uint32_t)(uintptr_t)(&tile[b][c * COL_STRIDE]) + k16 * 16;
            const char* g   = gbase + c * 256 + k16 * 16;
            asm volatile("global_load_async_to_lds_b128 %0, %1, off"
                         :: "v"(ldsoff), "v"(g) : "memory");
        }
    };

    issue_chunk(0, 0);
    asm volatile("s_wait_asynccnt 0x0" ::: "memory");
    __syncthreads();

    int p = 0;
    for (int cc = 0; cc < NCHUNK; ++cc) {
        if (cc + 1 < NCHUNK) issue_chunk(cc + 1, p ^ 1);  // overlap DMA w/ compute

        const int cbase = cc * CHUNK_COLS;
        #pragma unroll
        for (int ct = 0; ct < 4; ++ct) {
            // B fragment (32x16, f16): lane<16 holds K {0..7, 16..23} of its column,
            // lane>=16 holds K {8..15, 24..31} -> two 8-half chunks from LDS.
            const _Float16* brow =
                &tile[p][(ct * 16 + l15) * COL_STRIDE] + halfsel * 8;
            v8f c = {};
            #pragma unroll
            for (int kc = 0; kc < 4; ++kc) {
                v16h bfrag;
                *(float4*)&bfrag       = *(const float4*)(brow + kc * 32);
                *((float4*)&bfrag + 1) = *(const float4*)(brow + kc * 32 + 16);
                c = __builtin_amdgcn_wmma_f32_16x16x32_f16(
                        false, afrag[kc], false, bfrag,
                        (short)0, c, false, false);
            }
            const int col = cbase + ct * 16 + l15;
            #pragma unroll
            for (int v = 0; v < 8; ++v) {
                const int row = r0 + v + halfsel * 8;
                const float e = __expf(c[v] * INV_T);
                acc[v] += (row == col) ? 0.0f : e;   // exclude self-diagonal
            }
        }

        asm volatile("s_wait_asynccnt 0x0" ::: "memory");
        __syncthreads();
        p ^= 1;
    }

    // Reduce each row's partial sum across the 16 lanes of its half-wave.
    #pragma unroll
    for (int v = 0; v < 8; ++v) {
        float s = acc[v];
        #pragma unroll
        for (int m = 8; m >= 1; m >>= 1) s += __shfl_xor(s, m, 32);
        acc[v] = s;
    }

    if (l15 == 0) {   // lane 0 -> rows r0..r0+7, lane 16 -> rows r0+8..r0+15
        const int rbase = r0 + halfsel * 8;
        #pragma unroll
        for (int v = 0; v < 8; ++v)
            rowloss[rbase + v] = __logf(acc[v]) - pos[rbase + v];
    }
}

// ---------------------------------------------------------------------------
// Kernel 4: final reduction, loss = sum(rowloss) / N. Writes d_out[0].
// ---------------------------------------------------------------------------
__global__ void __launch_bounds__(256)
reduce_kernel(const float* __restrict__ rowloss, float* __restrict__ out) {
    __shared__ float sm[256];
    float s = 0.0f;
    for (int i = threadIdx.x; i < N_ROWS; i += 256) s += rowloss[i];
    sm[threadIdx.x] = s;
    __syncthreads();
    for (int m = 128; m >= 1; m >>= 1) {
        if ((int)threadIdx.x < m) sm[threadIdx.x] += sm[threadIdx.x + m];
        __syncthreads();
    }
    if (threadIdx.x == 0) out[0] = sm[0] / (float)N_ROWS;
}

// ---------------------------------------------------------------------------
extern "C" void kernel_launch(void* const* d_in, const int* in_sizes, int n_in,
                              void* d_out, int out_size, void* d_ws, size_t ws_size,
                              hipStream_t stream) {
    const float* zi = (const float*)d_in[0];
    const float* zj = (const float*)d_in[1];

    char* ws = (char*)d_ws;
    _Float16* znh   = (_Float16*)ws;                                 // 2 MB
    float*    pos   = (float*)(ws + (size_t)N_ROWS * D_DIM * 2);     // 32 KB
    float*    rloss = pos + N_ROWS;                                  // 32 KB
    float*    out   = (float*)d_out;

    normalize_kernel   <<<N_ROWS / 8,   256, 0, stream>>>(zi, zj, znh);
    pos_kernel         <<<N_ROWS / 8,   256, 0, stream>>>(znh, pos);
    gram_rowloss_kernel<<<N_ROWS / 128, 256, 0, stream>>>(znh, pos, rloss);
    reduce_kernel      <<<1,            256, 0, stream>>>(rloss, out);
}
